// HierarchicalEncoder_47055661695499
// MI455X (gfx1250) — matrix operand order repaired and checked by reading, not compiled, not run on hardware
//
#include <hip/hip_runtime.h>
#include <cstdint>
#include <cstddef>

// ---------------------------------------------------------------------------
// CDNA5 (gfx1250, wave32) hierarchical GRU encoder.
// Matrix work: v_wmma_f32_16x16x32_f16 (f16 A/B, f32 acc).
// Input staging: double-buffered GLOBAL_LOAD_ASYNC_TO_LDS_B128 (ASYNCcnt).
// ---------------------------------------------------------------------------

typedef __attribute__((ext_vector_type(16))) _Float16 v16h;
typedef __attribute__((ext_vector_type(8)))  float    v8f;

union FragU { uint4 u[2]; v16h h; };

#define BB     128   // batch
#define TT     50    // utterances per dialogue
#define WW     30    // tokens per utterance
#define INW    300   // word embedding dim
#define HH     128   // hidden
#define GATES  384   // 3*H
#define NTILES 24    // 384/16
#define KPAD_W 320   // 300 padded to multiple of 32
#define KC_H   4     // 128/32

// ---------------------------------------------------------------------------
// Weight pre-pack: W (f32, row-major (384, K)) -> f16 B-fragments in the
// per-lane contiguous-k order (same as gemm3x's A loads):
//   fragment index: ((tile*KC + kc)*32 + lane) * 8 dwords + j
//   lane: N = lane&15 (col = tile*16+N), khalf = lane>>4
//   dword j holds f16 pair (k0, k0+1), k0 = kc*32 + khalf*16 + 2*j
// ---------------------------------------------------------------------------
__global__ __launch_bounds__(256) void pack_w(const float* __restrict__ W,
                                              unsigned* __restrict__ out,
                                              int K, int KC) {
  const int total = NTILES * KC * 32 * 8;
  for (int idx = blockIdx.x * 256 + threadIdx.x; idx < total;
       idx += gridDim.x * 256) {
    const int j     = idx & 7;
    const int lane  = (idx >> 3) & 31;
    const int rest  = idx >> 8;
    const int kc    = rest % KC;
    const int nt    = rest / KC;
    const int n     = lane & 15;
    const int khalf = lane >> 4;
    const int col   = nt * 16 + n;
    const int k0    = kc * 32 + khalf * 16 + 2 * j;
    const _Float16 lo = (k0     < K) ? (_Float16)W[(size_t)col * K + k0]     : (_Float16)0.f;
    const _Float16 hi = (k0 + 1 < K) ? (_Float16)W[(size_t)col * K + k0 + 1] : (_Float16)0.f;
    const unsigned ul = __builtin_bit_cast(unsigned short, lo);
    const unsigned uh = __builtin_bit_cast(unsigned short, hi);
    out[idx] = ul | (uh << 16);
  }
}

struct GruParams {
  const float*    x_f32;     // word-level input (f32, strided rows)
  const _Float16* x_f16;     // persona/sentence input (f16, (T,B,256))
  int K, Kpad, KC_in;        // input-projection K dims
  const uint4* WiF;          // packed input-proj weights
  const uint4* WhF;          // packed recurrent weights
  const float* bi;
  const float* bh;
  const int*   P;            // reset flags (B,T) int32
  int reset_mode;            // 0 none, 1 d=P[b][t], 2 d=P[b][t+1] (last: 1)
  int nsteps;
  int reverse;
  int skip_gh;               // word backward: h==0, skip Wh GEMM
  int word_mode;             // word-level input/output mapping
  int x_step_off;            // word: starting token index
  _Float16* out_f16;
  float*    out_f32;
  int out_col_off;           // 0 = fwd half, 128 = bwd half
  int out_every;             // write each step vs only last
  float* tail;               // hs[-1] region (sentence) or null
};

// Async global -> LDS copy (tracked by ASYNCcnt). Low 32 bits of a generic
// __shared__ pointer == raw LDS offset (LDS aperture keeps offset in [31:0]).
__device__ __forceinline__ void async_b128(unsigned lds_off, uint64_t gaddr) {
  asm volatile("global_load_async_to_lds_b128 %0, %1, off"
               :: "v"(lds_off), "v"(gaddr) : "memory");
}
__device__ __forceinline__ void wait_async0() {
  asm volatile("s_wait_asynccnt 0x0" ::: "memory");
}

// A (16 x K) from LDS, B from packed global fragments, 3 output tiles / wave.
// A per-lane layout (16-bit A 16x32): m=lane&15, half=lane>>4:
//   dwords 0..3 : K = kc*32 + half*16 + {0..7}
//   dwords 4..7 : K = kc*32 + half*16 + 8 + {0..7}
__device__ __forceinline__ void gemm3x(v8f acc[3], const _Float16* A, int Kpad,
                                       int KC, const uint4* BF, int tile0,
                                       int lane) {
  const int m = lane & 15, half = lane >> 4;
  for (int kc = 0; kc < KC; ++kc) {
    FragU a;
    const _Float16* ap = A + m * Kpad + kc * 32 + half * 16;
    a.u[0] = *(const uint4*)(ap);
    a.u[1] = *(const uint4*)(ap + 8);
#pragma unroll
    for (int i = 0; i < 3; ++i) {
      FragU b;
      const uint4* bp = BF + ((size_t)((tile0 + i) * KC + kc) * 32 + lane) * 2;
      b.u[0] = bp[0];
      b.u[1] = bp[1];
      acc[i] = __builtin_amdgcn_wmma_f32_16x16x32_f16(
          false, a.h, false, b.h, (short)0, acc[i], false, false);
    }
  }
}

__global__ __launch_bounds__(256) void gru_wmma(GruParams p) {
  // word mode: async -> sXraw[buf] (f32), convert -> sAx[0] (f16, padded)
  // seq  mode: async -> sAx[buf] directly (already f16, contiguous 8KB/step)
  __shared__ __align__(16) _Float16 sAx[2][16 * KPAD_W];   // 20 KB
  __shared__ __align__(16) float    sXraw[2][16 * INW];    // 38.4 KB
  __shared__ __align__(16) _Float16 sAh[16 * HH];          // hidden (f16)
  __shared__ float sRZ[16 * 256];                          // r|z pre-act
  __shared__ float sIC[16 * 128];                          // ic pre-act
  __shared__ float sHC[16 * 128];                          // hc pre-act
  __shared__ float sBi[GATES];
  __shared__ float sBh[GATES];

  const int tid  = threadIdx.x;
  const int mrow = tid >> 4;    // 16 threads per row for staging
  const int li   = tid & 15;

  for (int i = tid; i < GATES; i += 256) { sBi[i] = p.bi[i]; sBh[i] = p.bh[i]; }
  for (int i = tid; i < 16 * HH; i += 256) sAh[i] = (_Float16)0.f;
  if (p.word_mode) {  // zero the K pad region once (cols 300..319 of sAx[0])
    for (int k = INW + li; k < KPAD_W; k += 16)
      sAx[0][mrow * KPAD_W + k] = (_Float16)0.f;
  }

  // issue async copy of the row data for a given time index into buffer `buf`
  auto issue = [&](int t, int buf) {
    if (p.word_mode) {
      const size_t s = (size_t)blockIdx.x * 16 + mrow;
      const uint64_t gbase =
          (uint64_t)(p.x_f32 + (s * WW + (size_t)(p.x_step_off + t)) * INW);
      const unsigned lbase = (unsigned)(uintptr_t)&sXraw[buf][mrow * INW];
#pragma unroll
      for (int j = 0; j < 5; ++j) {           // 75 x 16B chunks per row
        const int idx = li + j * 16;
        if (idx < 75) async_b128(lbase + idx * 16, gbase + (uint64_t)idx * 16);
      }
    } else {
      // rows for this block are contiguous: 16*512B = 8KB
      const uint64_t gbase = (uint64_t)(
          p.x_f16 + ((size_t)t * BB + (size_t)blockIdx.x * 16) * 256);
      const unsigned lbase = (unsigned)(uintptr_t)&sAx[buf][0];
#pragma unroll
      for (int j = 0; j < 2; ++j) {           // 512 x 16B chunks total
        const unsigned off = (unsigned)(tid + j * 256) * 16;
        async_b128(lbase + off, gbase + off);
      }
    }
  };

  const int lane  = tid & 31;
  const int wave  = tid >> 5;
  const int tile0 = wave * 3;

  int cur = 0;
  issue(p.reverse ? p.nsteps - 1 : 0, 0);   // prefetch step 0

  for (int it = 0; it < p.nsteps; ++it) {
    const int t = p.reverse ? (p.nsteps - 1 - it) : it;

    // ---- wait for this step's async copy; kick off the next one -----------
    wait_async0();
    if (it + 1 < p.nsteps)
      issue(p.reverse ? (t - 1) : (t + 1), cur ^ 1);

    // ---- phase A: convert (word) + hidden reset ---------------------------
    if (p.word_mode) {
      // row mrow was copied by this same wave -> per-wave asynccnt wait is
      // sufficient before reading it here.
      for (int k = li; k < INW; k += 16)
        sAx[0][mrow * KPAD_W + k] = (_Float16)sXraw[cur][mrow * INW + k];
    }
    if (p.reset_mode) {
      const int b = blockIdx.x * 16 + mrow;
      int d;
      if (p.reset_mode == 1) d = p.P[b * TT + t];
      else                   d = (t < p.nsteps - 1) ? p.P[b * TT + t + 1] : 1;
      if (d)
        for (int c = li; c < HH; c += 16) sAh[mrow * HH + c] = (_Float16)0.f;
    }
    __syncthreads();   // A tile + reset h visible to all waves

    // ---- phase B: WMMA GEMMs (gi = x@Wi^T, gh = h@Wh^T), f32 accum --------
    const _Float16* Abase = p.word_mode ? sAx[0] : sAx[cur];
    v8f zero = {0.f, 0.f, 0.f, 0.f, 0.f, 0.f, 0.f, 0.f};
    v8f accI[3] = {zero, zero, zero};
    v8f accH[3] = {zero, zero, zero};
    gemm3x(accI, Abase, p.Kpad, p.KC_in, p.WiF, tile0, lane);
    if (!p.skip_gh) gemm3x(accH, sAh, HH, KC_H, p.WhF, tile0, lane);

    // ---- phase C: stage pre-activations (+biases) to LDS ------------------
    {
      const int half = lane >> 4, n = lane & 15;
#pragma unroll
      for (int i = 0; i < 3; ++i) {
        const int nt = tile0 + i, col = nt * 16 + n;
        const float bI = sBi[col], bH = sBh[col];
#pragma unroll
        for (int v = 0; v < 8; ++v) {
          const int m = half * 8 + v;  // C/D layout: VGPR v, lane-half -> M
          if (nt < 16) {
            sRZ[m * 256 + col] = accI[i][v] + accH[i][v] + bI + bH;
          } else {
            const int c = col - 256;
            sIC[m * 128 + c] = accI[i][v] + bI;
            sHC[m * 128 + c] = accH[i][v] + bH;
          }
        }
      }
    }
    __syncthreads();

    // ---- phase D: gates, hidden update, output ----------------------------
    const bool last = (it == p.nsteps - 1);
    for (int idx = tid; idx < 16 * 128; idx += 256) {
      const int m = idx >> 7, c = idx & 127;
      const float r     = 1.f / (1.f + __expf(-sRZ[m * 256 + c]));
      const float zg    = 1.f / (1.f + __expf(-sRZ[m * 256 + 128 + c]));
      const float ng    = tanhf(sIC[idx] + r * sHC[idx]);
      const float hprev = (float)sAh[idx];
      const float hn    = (1.f - zg) * ng + zg * hprev;
      sAh[idx] = (_Float16)hn;
      if (p.out_every || last) {
        size_t orow;
        if (p.word_mode) {
          const size_t s  = (size_t)blockIdx.x * 16 + m;
          const size_t bb = s / TT, tt = s % TT;   // s = b*T + t
          orow = (tt * BB + bb) * 256;
        } else {
          const size_t b = (size_t)blockIdx.x * 16 + m;
          orow = ((size_t)t * BB + b) * 256;
        }
        const size_t o = orow + p.out_col_off + c;
        if (p.out_f16) p.out_f16[o] = (_Float16)hn;
        else           p.out_f32[o] = hn;
        if (p.tail && t == TT - 1)
          p.tail[((size_t)blockIdx.x * 16 + m) * 256 + p.out_col_off + c] = hn;
      }
    }
    __syncthreads();
    cur ^= 1;
  }
}

// ---------------------------------------------------------------------------
// Host side
// ---------------------------------------------------------------------------
extern "C" void kernel_launch(void* const* d_in, const int* in_sizes, int n_in,
                              void* d_out, int out_size, void* d_ws,
                              size_t ws_size, hipStream_t stream) {
  (void)in_sizes; (void)n_in; (void)out_size; (void)ws_size;

  const float* X = (const float*)d_in[0];
  const int*   P = (const int*)d_in[1];

  char* ws = (char*)d_ws;
  size_t off = 0;
  auto take = [&](size_t bytes) { void* q = ws + off; off += bytes; return q; };

  // packed fragment buffers (bytes = 24 * KC * 32 * 8 * 4)
  uint32_t* pk_wWiF = (uint32_t*)take(245760);  // K=300->320, KC=10
  uint32_t* pk_wWiB = (uint32_t*)take(245760);
  uint32_t* pk_wWhF = (uint32_t*)take(98304);   // K=128, KC=4
  uint32_t* pk_pWiF = (uint32_t*)take(196608);  // K=256, KC=8
  uint32_t* pk_pWhF = (uint32_t*)take(98304);
  uint32_t* pk_pWiB = (uint32_t*)take(196608);
  uint32_t* pk_pWhB = (uint32_t*)take(98304);
  uint32_t* pk_sWiF = (uint32_t*)take(196608);
  uint32_t* pk_sWhF = (uint32_t*)take(98304);
  uint32_t* pk_sWiB = (uint32_t*)take(196608);
  uint32_t* pk_sWhB = (uint32_t*)take(98304);
  _Float16* hw = (_Float16*)take((size_t)TT * BB * 256 * 2);  // word output
  _Float16* hp = (_Float16*)take((size_t)TT * BB * 256 * 2);  // persona output

  auto pack = [&](const void* W, uint32_t* out, int K, int KC) {
    const int total = NTILES * KC * 32 * 8;
    pack_w<<<(total + 255) / 256, 256, 0, stream>>>((const float*)W, out, K, KC);
  };
  pack(d_in[2],  pk_wWiF, 300, 10);
  pack(d_in[3],  pk_wWhF, 128, 4);
  pack(d_in[6],  pk_wWiB, 300, 10);   // w_Wh_b unused (h==0 for bwd cell)
  pack(d_in[10], pk_pWiF, 256, 8);
  pack(d_in[11], pk_pWhF, 128, 4);
  pack(d_in[14], pk_pWiB, 256, 8);
  pack(d_in[15], pk_pWhB, 128, 4);
  pack(d_in[18], pk_sWiF, 256, 8);
  pack(d_in[19], pk_sWhF, 128, 4);
  pack(d_in[22], pk_sWiB, 256, 8);
  pack(d_in[23], pk_sWhB, 128, 4);

  // ---- word level, forward: 6400 sequences, 30 steps ----------------------
  GruParams q{};
  q.x_f32 = X; q.x_f16 = nullptr;
  q.K = 300; q.Kpad = KPAD_W; q.KC_in = 10;
  q.WiF = (const uint4*)pk_wWiF; q.WhF = (const uint4*)pk_wWhF;
  q.bi = (const float*)d_in[4]; q.bh = (const float*)d_in[5];
  q.P = nullptr; q.reset_mode = 0;
  q.nsteps = WW; q.reverse = 0; q.skip_gh = 0;
  q.word_mode = 1; q.x_step_off = 0;
  q.out_f16 = hw; q.out_f32 = nullptr; q.out_col_off = 0; q.out_every = 0;
  q.tail = nullptr;
  gru_wmma<<<400, 256, 0, stream>>>(q);

  // ---- word level, backward: single cell on x[W-1] with h=0 ---------------
  q.WiF = (const uint4*)pk_wWiB;
  q.bi = (const float*)d_in[8]; q.bh = (const float*)d_in[9];
  q.nsteps = 1; q.skip_gh = 1; q.x_step_off = WW - 1; q.out_col_off = 128;
  gru_wmma<<<400, 256, 0, stream>>>(q);

  // ---- persona level (DiscontinuedGRU), forward ---------------------------
  GruParams r{};
  r.x_f32 = nullptr; r.x_f16 = hw;
  r.K = 256; r.Kpad = 256; r.KC_in = 8;
  r.WiF = (const uint4*)pk_pWiF; r.WhF = (const uint4*)pk_pWhF;
  r.bi = (const float*)d_in[12]; r.bh = (const float*)d_in[13];
  r.P = P; r.reset_mode = 1;
  r.nsteps = TT; r.reverse = 0; r.skip_gh = 0; r.word_mode = 0; r.x_step_off = 0;
  r.out_f16 = hp; r.out_f32 = nullptr; r.out_col_off = 0; r.out_every = 1;
  r.tail = nullptr;
  gru_wmma<<<8, 256, 0, stream>>>(r);

  // ---- persona level, backward (reset uses D[t+1], last step = 1) ---------
  r.WiF = (const uint4*)pk_pWiB; r.WhF = (const uint4*)pk_pWhB;
  r.bi = (const float*)d_in[16]; r.bh = (const float*)d_in[17];
  r.reset_mode = 2; r.reverse = 1; r.out_col_off = 128;
  gru_wmma<<<8, 256, 0, stream>>>(r);

  // ---- sentence level, forward (writes f32 to d_out + tail) ---------------
  GruParams s{};
  s.x_f32 = nullptr; s.x_f16 = hp;
  s.K = 256; s.Kpad = 256; s.KC_in = 8;
  s.WiF = (const uint4*)pk_sWiF; s.WhF = (const uint4*)pk_sWhF;
  s.bi = (const float*)d_in[20]; s.bh = (const float*)d_in[21];
  s.P = nullptr; s.reset_mode = 0;
  s.nsteps = TT; s.reverse = 0; s.skip_gh = 0; s.word_mode = 0; s.x_step_off = 0;
  s.out_f16 = nullptr; s.out_f32 = (float*)d_out;
  s.out_col_off = 0; s.out_every = 1;
  s.tail = (float*)d_out + (size_t)TT * BB * 256;
  gru_wmma<<<8, 256, 0, stream>>>(s);

  // ---- sentence level, backward -------------------------------------------
  s.WiF = (const uint4*)pk_sWiB; s.WhF = (const uint4*)pk_sWhB;
  s.bi = (const float*)d_in[24]; s.bh = (const float*)d_in[25];
  s.reverse = 1; s.out_col_off = 128;
  gru_wmma<<<8, 256, 0, stream>>>(s);
}
